// ChaoticEvolutionGNN_67946382623035
// MI455X (gfx1250) — matrix operand order repaired and checked by reading, not compiled, not run on hardware
//
#include <hip/hip_runtime.h>

// ---------------- constants ----------------
constexpr int NN     = 10000;
constexpr int NP     = 10016;  // padded to multiple of 32 for 32-row GEMM tiles
constexpr int EE     = 160000;
constexpr int F_IN   = 128;
constexpr int HH     = 64;     // hidden
constexpr int HEADS  = 8;
constexpr int LL     = 6;
constexpr int MDIM   = 512;    // HEADS*HH
#define EPSF 1e-5f

// ---------------- WMMA types ----------------
typedef __attribute__((ext_vector_type(16))) __bf16 v16bf;
typedef __attribute__((ext_vector_type(8)))  float  v8f;

union Frag32B { uint4 u[2]; v16bf v; };

// ---------------- helpers ----------------
__device__ __forceinline__ unsigned short f2bf(float f) {
    unsigned u = __float_as_uint(f);
    unsigned r = u + 0x7FFFu + ((u >> 16) & 1u);
    return (unsigned short)(r >> 16);
}

__device__ __forceinline__ unsigned f2ord(float f) {
    unsigned u = __float_as_uint(f);
    return (u & 0x80000000u) ? ~u : (u | 0x80000000u);
}
__device__ __forceinline__ float ord2f(unsigned e) {
    return __uint_as_float((e & 0x80000000u) ? (e ^ 0x80000000u) : ~e);
}

__device__ __forceinline__ float hash_normal(unsigned seed, unsigned idx) {
    unsigned s = seed * 0x9E3779B9u ^ idx * 0x85EBCA6Bu;
    s ^= s >> 16; s *= 0x7FEB352Du; s ^= s >> 15; s *= 0x846CA68Bu; s ^= s >> 16;
    unsigned s2 = s * 0x27D4EB2Fu + 0x165667B1u;
    s2 ^= s2 >> 15; s2 *= 0x2C1B3C6Du; s2 ^= s2 >> 12;
    float u1 = (float)(s  & 0xFFFFFFu) * (1.0f / 16777216.0f) + 1e-7f;
    float u2 = (float)(s2 & 0xFFFFFFu) * (1.0f / 16777216.0f);
    return sqrtf(-2.0f * logf(u1)) * __cosf(6.28318530718f * u2);
}

template <int BS>
__device__ __forceinline__ float blockSum(float v, float* sh) {
    int t = threadIdx.x;
    sh[t] = v; __syncthreads();
#pragma unroll
    for (int s = BS / 2; s > 0; s >>= 1) {
        if (t < s) sh[t] += sh[t + s];
        __syncthreads();
    }
    float r = sh[0]; __syncthreads();
    return r;
}

// ---------------- WMMA GEMM: C[M,Nc] = A[M,K] * Bt[Nc,K]^T ----------------
// A row-major bf16 (M multiple of 32), Bt row-major bf16 (pre-transposed weight),
// C f32 row-major. One wave -> 32x64 C block: 2 A-frags x 4 B-frags = 8 WMMAs
// per K step of 32 (B traffic amortized over 32 output rows).
__global__ __launch_bounds__(256)
void k_gemm_bf16(const unsigned short* __restrict__ A,
                 const unsigned short* __restrict__ Bt,
                 float* __restrict__ C, int M, int K, int Nc) {
    const int lane = threadIdx.x & 31;
    const int wid  = threadIdx.x >> 5;
    const int colStrips = Nc >> 6;
    const int strip = blockIdx.x * 8 + wid;
    const int rowTile = strip / colStrips;
    if (rowTile * 32 >= M) return;
    const int colStrip = strip % colStrips;
    const int row0 = rowTile * 32;
    const int col0 = colStrip * 64;
    const int half = lane >> 4;      // 0: lanes 0-15, 1: lanes 16-31
    const int l16  = lane & 15;

    v8f accA0 = {}, accA1 = {}, accA2 = {}, accA3 = {};
    v8f accB0 = {}, accB1 = {}, accB2 = {}, accB3 = {};

    // A fragment 16x32: lanes<16 hold K {0..7,16..23}; lanes>=16 hold {8..15,24..31}
    const unsigned short* aPtr0 = A + (size_t)(row0 + l16) * K;
    const unsigned short* aPtr1 = aPtr0 + (size_t)16 * K;
    const int aOff0 = half * 8;
    const int aOff1 = 16 + half * 8;
    // B fragment 32x16 via Bt rows: lane column = l16, K {0..7,16..23} (half=0) / {8..15,24..31} (half=1)
    const unsigned short* bPtr = Bt + (size_t)(col0 + l16) * K;

    for (int k0 = 0; k0 < K; k0 += 32) {
        Frag32B a0, a1, b0, b1, b2, b3;
        a0.u[0] = *(const uint4*)(aPtr0 + k0 + aOff0);
        a0.u[1] = *(const uint4*)(aPtr0 + k0 + aOff1);
        a1.u[0] = *(const uint4*)(aPtr1 + k0 + aOff0);
        a1.u[1] = *(const uint4*)(aPtr1 + k0 + aOff1);
        const unsigned short* bb = bPtr + k0;
        b0.u[0] = *(const uint4*)(bb + aOff0);
        b0.u[1] = *(const uint4*)(bb + aOff1);
        b1.u[0] = *(const uint4*)(bb + (size_t)16 * K + aOff0);
        b1.u[1] = *(const uint4*)(bb + (size_t)16 * K + aOff1);
        b2.u[0] = *(const uint4*)(bb + (size_t)32 * K + aOff0);
        b2.u[1] = *(const uint4*)(bb + (size_t)32 * K + aOff1);
        b3.u[0] = *(const uint4*)(bb + (size_t)48 * K + aOff0);
        b3.u[1] = *(const uint4*)(bb + (size_t)48 * K + aOff1);
        if (k0 + 32 < K) {
            __builtin_prefetch(aPtr0 + k0 + 32, 0, 1);
            __builtin_prefetch(aPtr1 + k0 + 32, 0, 1);
        }

        accA0 = __builtin_amdgcn_wmma_f32_16x16x32_bf16(false, a0.v, false, b0.v, (short)0, accA0, false, false);
        accA1 = __builtin_amdgcn_wmma_f32_16x16x32_bf16(false, a0.v, false, b1.v, (short)0, accA1, false, false);
        accA2 = __builtin_amdgcn_wmma_f32_16x16x32_bf16(false, a0.v, false, b2.v, (short)0, accA2, false, false);
        accA3 = __builtin_amdgcn_wmma_f32_16x16x32_bf16(false, a0.v, false, b3.v, (short)0, accA3, false, false);
        accB0 = __builtin_amdgcn_wmma_f32_16x16x32_bf16(false, a1.v, false, b0.v, (short)0, accB0, false, false);
        accB1 = __builtin_amdgcn_wmma_f32_16x16x32_bf16(false, a1.v, false, b1.v, (short)0, accB1, false, false);
        accB2 = __builtin_amdgcn_wmma_f32_16x16x32_bf16(false, a1.v, false, b2.v, (short)0, accB2, false, false);
        accB3 = __builtin_amdgcn_wmma_f32_16x16x32_bf16(false, a1.v, false, b3.v, (short)0, accB3, false, false);
    }

    // C/D layout: element r -> row = rowBase + half*8 + r, col = col0 + l16 (+16*c)
    float* cPtr0 = C + (size_t)(row0 + half * 8) * Nc + col0 + l16;
    float* cPtr1 = cPtr0 + (size_t)16 * Nc;
#pragma unroll
    for (int r = 0; r < 8; ++r) {
        cPtr0[(size_t)r * Nc + 0]  = accA0[r];
        cPtr0[(size_t)r * Nc + 16] = accA1[r];
        cPtr0[(size_t)r * Nc + 32] = accA2[r];
        cPtr0[(size_t)r * Nc + 48] = accA3[r];
        cPtr1[(size_t)r * Nc + 0]  = accB0[r];
        cPtr1[(size_t)r * Nc + 16] = accB1[r];
        cPtr1[(size_t)r * Nc + 32] = accB2[r];
        cPtr1[(size_t)r * Nc + 48] = accB3[r];
    }
}

// ---------------- conversion / transpose ----------------
__global__ void k_f32_to_bf16(const float* __restrict__ s, unsigned short* __restrict__ d, int n) {
    int t = blockIdx.x * blockDim.x + threadIdx.x;
    if (t < n) d[t] = f2bf(s[t]);
}

// W[K,Nc] f32 -> Wt[Nc,K] bf16
__global__ void k_transpose_bf16(const float* __restrict__ W, unsigned short* __restrict__ Wt, int K, int Nc) {
    int t = blockIdx.x * blockDim.x + threadIdx.x;
    if (t >= K * Nc) return;
    int k = t / Nc, n = t % Nc;
    Wt[(size_t)n * K + k] = f2bf(W[t]);
}

__global__ void k_fill_u32(unsigned* __restrict__ p, unsigned v, int n) {
    int t = blockIdx.x * blockDim.x + threadIdx.x;
    if (t < n) p[t] = v;
}

// ---------------- encoder epilogue: bias + LN(g,b) + relu + chaos ----------------
__global__ void k_enc_post(const float* __restrict__ t, const float* __restrict__ b,
                           const float* __restrict__ g, const float* __restrict__ be,
                           const float* __restrict__ chaos,
                           float* __restrict__ h, unsigned short* __restrict__ hbf) {
    __shared__ float sh[64];
    int n = blockIdx.x, c = threadIdx.x;
    float v  = t[(size_t)n * 64 + c] + b[c];
    float mu = blockSum<64>(v, sh) * (1.0f / 64.0f);
    float dv = v - mu;
    float var = blockSum<64>(dv * dv, sh) * (1.0f / 64.0f);
    float y = dv * rsqrtf(var + EPSF) * g[c] + be[c];
    y = fmaxf(y, 0.0f) + chaos[c] * 0.1f;
    h[(size_t)n * 64 + c]   = y;
    hbf[(size_t)n * 64 + c] = f2bf(y);
}

// ---------------- fractal transform epilogue (width 512) ----------------
__global__ void k_tr_post(const float* __restrict__ t, const float* __restrict__ b,
                          const float* __restrict__ g, const float* __restrict__ be,
                          float* __restrict__ feats, unsigned short* __restrict__ curbf,
                          int d, float cf, int seed) {
    __shared__ float sh[256];
    int n = blockIdx.x, tx = threadIdx.x;
    int c0 = tx, c1 = tx + 256;
    size_t i0 = (size_t)n * 512 + c0, i1 = i0 + 256;
    float v0 = t[i0] + b[c0];
    float v1 = t[i1] + b[c1];
    float mu = blockSum<256>(v0 + v1, sh) * (1.0f / 512.0f);
    float d0 = v0 - mu, d1 = v1 - mu;
    float var = blockSum<256>(d0 * d0 + d1 * d1, sh) * (1.0f / 512.0f);
    float inv = rsqrtf(var + EPSF);
    float y0 = fmaxf(d0 * inv * g[c0] + be[c0], 0.0f);
    float y1 = fmaxf(d1 * inv * g[c1] + be[c1], 0.0f);
    if (d == 0) { feats[i0] = y0; feats[i1] = y1; }
    else        { feats[i0] += y0; feats[i1] += y1; }
    if (d < 2) {
        curbf[i0] = f2bf(y0 + hash_normal((unsigned)seed, (unsigned)i0) * cf);
        curbf[i1] = f2bf(y1 + hash_normal((unsigned)seed, (unsigned)i1) * cf);
    }
}

// ---------------- attention scores ----------------
__global__ void k_scores(const float* __restrict__ feats, const float* __restrict__ att,
                         float* __restrict__ ai, float* __restrict__ aj) {
    int t = blockIdx.x * blockDim.x + threadIdx.x;
    if (t >= NN * HEADS) return;
    int n = t >> 3, hh = t & 7;
    const float* f  = feats + (size_t)n * MDIM + hh * HH;
    const float* aI = att + hh * (2 * HH);
    const float* aJ = aI + HH;
    float si = 0.f, sj = 0.f;
#pragma unroll 8
    for (int c = 0; c < HH; ++c) {
        float fv = f[c] * (1.0f / 3.0f);
        si += fv * aI[c];
        sj += fv * aJ[c];
    }
    ai[t] = si; aj[t] = sj;
}

// ---------------- edge passes ----------------
__global__ void k_edge1(const int* __restrict__ ei, const float* __restrict__ ai,
                        const float* __restrict__ aj, float* __restrict__ alpha,
                        unsigned* __restrict__ amax) {
    int e = blockIdx.x * blockDim.x + threadIdx.x;
    if (e >= EE) return;
    int s = ei[e], dd = ei[EE + e];
#pragma unroll
    for (int hh = 0; hh < HEADS; ++hh) {
        float al = ai[dd * 8 + hh] + aj[s * 8 + hh];
        al = al > 0.f ? al : 0.2f * al;
        alpha[(size_t)e * 8 + hh] = al;
        atomicMax(&amax[dd * 8 + hh], f2ord(al));
    }
}

__global__ void k_edge2(const int* __restrict__ ei, float* __restrict__ alpha,
                        const unsigned* __restrict__ amax, float* __restrict__ denom) {
    int e = blockIdx.x * blockDim.x + threadIdx.x;
    if (e >= EE) return;
    int dd = ei[EE + e];
#pragma unroll
    for (int hh = 0; hh < HEADS; ++hh) {
        float ex = __expf(alpha[(size_t)e * 8 + hh] - ord2f(amax[dd * 8 + hh]));
        alpha[(size_t)e * 8 + hh] = ex;
        atomicAdd(&denom[dd * 8 + hh], ex);
    }
}

__global__ void k_edge3(const int* __restrict__ ei, const float* __restrict__ ex,
                        const float* __restrict__ denom, const float* __restrict__ feats,
                        float* __restrict__ agg) {
    int t = blockIdx.x * blockDim.x + threadIdx.x;
    if (t >= EE * 64) return;
    int e = t >> 6, c = t & 63;
    int s = ei[e], dd = ei[EE + e];
#pragma unroll
    for (int hh = 0; hh < HEADS; ++hh) {
        float w = ex[(size_t)e * 8 + hh] / denom[dd * 8 + hh];
        atomicAdd(&agg[(size_t)dd * MDIM + hh * HH + c],
                  feats[(size_t)s * MDIM + hh * HH + c] * (1.0f / 3.0f) * w);
    }
}

// ---------------- head mean ----------------
__global__ void k_hnew(const float* __restrict__ agg, float* __restrict__ hnew,
                       unsigned short* __restrict__ hnewbf) {
    int t = blockIdx.x * blockDim.x + threadIdx.x;
    if (t >= NN * HH) return;
    int n = t >> 6, c = t & 63;
    float s = 0.f;
#pragma unroll
    for (int hh = 0; hh < HEADS; ++hh) s += agg[(size_t)n * MDIM + hh * HH + c];
    s *= 0.125f;
    hnew[t] = s; hnewbf[t] = f2bf(s);
}

// ---------------- evolution gate ----------------
__global__ void k_evo1post(const float* __restrict__ z, const float* __restrict__ b1,
                           unsigned short* __restrict__ z1bf) {
    int t = blockIdx.x * blockDim.x + threadIdx.x;
    if (t >= NN * 128) return;
    int c = t & 127;
    z1bf[t] = f2bf(fmaxf(z[t] + b1[c], 0.0f));
}

__global__ void k_gate(const float* __restrict__ z, const float* __restrict__ b2,
                       const float* __restrict__ hnew, const float* __restrict__ hprev,
                       float* __restrict__ hr, float* __restrict__ res) {
    int t = blockIdx.x * blockDim.x + threadIdx.x;
    if (t >= NN * HH) return;
    int c = t & 63;
    float zz = tanhf(z[t] + b2[c]);
    float g  = 1.0f / (1.0f + __expf(-zz));
    float hv = g * hnew[t] + (1.0f - g) * hprev[t];
    hr[t] = hv;
    res[t] += hv;
}

__global__ void k_ln_plain(const float* __restrict__ hr, float* __restrict__ h,
                           unsigned short* __restrict__ hbf) {
    __shared__ float sh[64];
    int n = blockIdx.x, c = threadIdx.x;
    float v  = hr[(size_t)n * 64 + c];
    float mu = blockSum<64>(v, sh) * (1.0f / 64.0f);
    float dv = v - mu;
    float var = blockSum<64>(dv * dv, sh) * (1.0f / 64.0f);
    float y = dv * rsqrtf(var + EPSF);
    h[(size_t)n * 64 + c]   = y;
    hbf[(size_t)n * 64 + c] = f2bf(y);
}

// ---------------- final decoders ----------------
__global__ void k_final(const float* __restrict__ res,
                        const float* __restrict__ sW, const float* __restrict__ sb,
                        const float* __restrict__ iW, const float* __restrict__ ib,
                        const float* __restrict__ cW, const float* __restrict__ cb,
                        const float* __restrict__ eW, const float* __restrict__ eb,
                        float* __restrict__ out) {
    __shared__ float hm[64];
    int n = blockIdx.x, c = threadIdx.x;
    float v = res[(size_t)n * 64 + c] * (1.0f / 6.0f);
    hm[c] = v;
    __syncthreads();
    float s = 0.f;
#pragma unroll 8
    for (int k = 0; k < 64; ++k) s += hm[k] * eW[k * 64 + c];
    out[(size_t)3 * NN + (size_t)n * 64 + c]  = s + eb[c];   // evolution
    out[(size_t)67 * NN + (size_t)n * 64 + c] = v;           // h_multi
    if (c < 3) {
        const float* W = (c == 0) ? sW : ((c == 1) ? iW : cW);
        float bb = (c == 0) ? sb[0] : ((c == 1) ? ib[0] : cb[0]);
        float acc = 0.f;
        for (int k = 0; k < 64; ++k) acc += hm[k] * W[k];
        out[(size_t)c * NN + n] = acc + bb;
    }
}

// ---------------- host ----------------
extern "C" void kernel_launch(void* const* d_in, const int* in_sizes, int n_in,
                              void* d_out, int out_size, void* d_ws, size_t ws_size,
                              hipStream_t stream) {
    (void)in_sizes; (void)n_in; (void)out_size; (void)ws_size;

    const float* x        = (const float*)d_in[0];
    const int*   ei       = (const int*)d_in[1];
    const float* enc_W    = (const float*)d_in[2];
    const float* enc_b    = (const float*)d_in[3];
    const float* enc_g    = (const float*)d_in[4];
    const float* enc_beta = (const float*)d_in[5];
    const float* chaos_m  = (const float*)d_in[6];
    const float* tW0 = (const float*)d_in[7];
    const float* tb0 = (const float*)d_in[8];
    const float* tg0 = (const float*)d_in[9];
    const float* tbe0 = (const float*)d_in[10];
    const float* tW1 = (const float*)d_in[11];
    const float* tb1 = (const float*)d_in[12];
    const float* tg1 = (const float*)d_in[13];
    const float* tbe1 = (const float*)d_in[14];
    const float* tW2 = (const float*)d_in[15];
    const float* tb2 = (const float*)d_in[16];
    const float* tg2 = (const float*)d_in[17];
    const float* tbe2 = (const float*)d_in[18];
    const float* att   = (const float*)d_in[19];
    const float* evoW1 = (const float*)d_in[20];
    const float* evob1 = (const float*)d_in[21];
    const float* evoW2 = (const float*)d_in[22];
    const float* evob2 = (const float*)d_in[23];
    const float* dsW = (const float*)d_in[24];
    const float* dsb = (const float*)d_in[25];
    const float* diW = (const float*)d_in[26];
    const float* dib = (const float*)d_in[27];
    const float* dcW = (const float*)d_in[28];
    const float* dcb = (const float*)d_in[29];
    const float* deW = (const float*)d_in[30];
    const float* deb = (const float*)d_in[31];
    float* out = (float*)d_out;

    // ---- workspace carve (GEMM operands/outputs padded to NP rows) ----
    char* p = (char*)d_ws;
    auto alloc = [&](size_t bytes) -> void* {
        void* r = (void*)p;
        p += (bytes + 255) & ~(size_t)255;
        return r;
    };
    unsigned short* x_bf    = (unsigned short*)alloc((size_t)NP * F_IN * 2);
    unsigned short* h_bf    = (unsigned short*)alloc((size_t)NP * HH * 2);
    unsigned short* cur_bf  = (unsigned short*)alloc((size_t)NP * MDIM * 2);
    unsigned short* hnew_bf = (unsigned short*)alloc((size_t)NP * HH * 2);
    unsigned short* z1_bf   = (unsigned short*)alloc((size_t)NP * 128 * 2);
    unsigned short* encWt   = (unsigned short*)alloc((size_t)HH * F_IN * 2);
    unsigned short* tW0t    = (unsigned short*)alloc((size_t)LL * MDIM * HH * 2);
    unsigned short* tW1t    = (unsigned short*)alloc((size_t)LL * MDIM * MDIM * 2);
    unsigned short* tW2t    = (unsigned short*)alloc((size_t)LL * MDIM * MDIM * 2);
    unsigned short* evoW1t  = (unsigned short*)alloc((size_t)128 * HH * 2);
    unsigned short* evoW2t  = (unsigned short*)alloc((size_t)HH * 128 * 2);
    float* h_f32  = (float*)alloc((size_t)NN * HH * 4);
    float* t_gemm = (float*)alloc((size_t)NP * MDIM * 4);   // aliased as agg
    float* feats  = (float*)alloc((size_t)NN * MDIM * 4);
    float* a_i    = (float*)alloc((size_t)NN * HEADS * 4);
    float* a_j    = (float*)alloc((size_t)NN * HEADS * 4);
    unsigned* amax = (unsigned*)alloc((size_t)NN * HEADS * 4);
    float* denom  = (float*)alloc((size_t)NN * HEADS * 4);
    float* alpha  = (float*)alloc((size_t)EE * HEADS * 4);
    float* h_new  = (float*)alloc((size_t)NN * HH * 4);
    float* hr     = (float*)alloc((size_t)NN * HH * 4);
    float* res    = (float*)alloc((size_t)NN * HH * 4);
    float* agg    = t_gemm;

    auto cdiv = [](int a, int b) { return (a + b - 1) / b; };
    auto gemm = [&](const unsigned short* A, const unsigned short* Bt, float* C, int K, int Nc) {
        int strips = (NP / 32) * (Nc / 64);
        k_gemm_bf16<<<cdiv(strips, 8), 256, 0, stream>>>(A, Bt, C, NP, K, Nc);
    };

    // ---- zero the 16 pad rows of each GEMM A operand (once per launch) ----
    const int PADR = NP - NN;  // 16
    k_fill_u32<<<cdiv(PADR * F_IN / 2, 256), 256, 0, stream>>>((unsigned*)(x_bf    + (size_t)NN * F_IN), 0u, PADR * F_IN / 2);
    k_fill_u32<<<cdiv(PADR * HH / 2,  256), 256, 0, stream>>>((unsigned*)(h_bf    + (size_t)NN * HH),   0u, PADR * HH / 2);
    k_fill_u32<<<cdiv(PADR * MDIM / 2,256), 256, 0, stream>>>((unsigned*)(cur_bf  + (size_t)NN * MDIM), 0u, PADR * MDIM / 2);
    k_fill_u32<<<cdiv(PADR * HH / 2,  256), 256, 0, stream>>>((unsigned*)(hnew_bf + (size_t)NN * HH),   0u, PADR * HH / 2);
    k_fill_u32<<<cdiv(PADR * 128 / 2, 256), 256, 0, stream>>>((unsigned*)(z1_bf   + (size_t)NN * 128),  0u, PADR * 128 / 2);

    // ---- one-time per launch: conversions / transposes ----
    k_f32_to_bf16<<<cdiv(NN * F_IN, 256), 256, 0, stream>>>(x, x_bf, NN * F_IN);
    k_transpose_bf16<<<cdiv(F_IN * HH, 256), 256, 0, stream>>>(enc_W, encWt, F_IN, HH);
    for (int i = 0; i < LL; ++i) {
        k_transpose_bf16<<<cdiv(HH * MDIM, 256), 256, 0, stream>>>(
            tW0 + (size_t)i * HH * MDIM, tW0t + (size_t)i * MDIM * HH, HH, MDIM);
        k_transpose_bf16<<<cdiv(MDIM * MDIM, 256), 256, 0, stream>>>(
            tW1 + (size_t)i * MDIM * MDIM, tW1t + (size_t)i * MDIM * MDIM, MDIM, MDIM);
        k_transpose_bf16<<<cdiv(MDIM * MDIM, 256), 256, 0, stream>>>(
            tW2 + (size_t)i * MDIM * MDIM, tW2t + (size_t)i * MDIM * MDIM, MDIM, MDIM);
    }
    k_transpose_bf16<<<cdiv(HH * 128, 256), 256, 0, stream>>>(evoW1, evoW1t, HH, 128);
    k_transpose_bf16<<<cdiv(128 * HH, 256), 256, 0, stream>>>(evoW2, evoW2t, 128, HH);

    k_fill_u32<<<cdiv(NN * HH, 256), 256, 0, stream>>>((unsigned*)res, 0u, NN * HH);

    // ---- encoder ----
    gemm(x_bf, encWt, t_gemm, F_IN, HH);
    k_enc_post<<<NN, 64, 0, stream>>>(t_gemm, enc_b, enc_g, enc_beta, chaos_m, h_f32, h_bf);

    const unsigned NEG_INF_ORD = 0x007FFFFFu;  // f2ord(-inf)

    // ---- layers ----
    for (int i = 0; i < LL; ++i) {
        float cf = 0.1f * (1.0f + 0.1f * (float)i);
        // fractal transform stack (WMMA GEMMs)
        gemm(h_bf, tW0t + (size_t)i * MDIM * HH, t_gemm, HH, MDIM);
        k_tr_post<<<NN, 256, 0, stream>>>(t_gemm, tb0 + i * MDIM, tg0 + i * MDIM,
                                          tbe0 + i * MDIM, feats, cur_bf, 0, cf, i * 10 + 0);
        gemm(cur_bf, tW1t + (size_t)i * MDIM * MDIM, t_gemm, MDIM, MDIM);
        k_tr_post<<<NN, 256, 0, stream>>>(t_gemm, tb1 + i * MDIM, tg1 + i * MDIM,
                                          tbe1 + i * MDIM, feats, cur_bf, 1, cf, i * 10 + 1);
        gemm(cur_bf, tW2t + (size_t)i * MDIM * MDIM, t_gemm, MDIM, MDIM);
        k_tr_post<<<NN, 256, 0, stream>>>(t_gemm, tb2 + i * MDIM, tg2 + i * MDIM,
                                          tbe2 + i * MDIM, feats, cur_bf, 2, cf, i * 10 + 2);

        // attention scores
        k_scores<<<cdiv(NN * HEADS, 256), 256, 0, stream>>>(feats, att + (size_t)i * HEADS * 2 * HH, a_i, a_j);

        // segment softmax + aggregation
        k_fill_u32<<<cdiv(NN * HEADS, 256), 256, 0, stream>>>(amax, NEG_INF_ORD, NN * HEADS);
        k_fill_u32<<<cdiv(NN * HEADS, 256), 256, 0, stream>>>((unsigned*)denom, 0u, NN * HEADS);
        k_edge1<<<cdiv(EE, 256), 256, 0, stream>>>(ei, a_i, a_j, alpha, amax);
        k_edge2<<<cdiv(EE, 256), 256, 0, stream>>>(ei, alpha, amax, denom);
        k_fill_u32<<<cdiv(NN * MDIM, 256), 256, 0, stream>>>((unsigned*)agg, 0u, NN * MDIM);
        k_edge3<<<cdiv(EE * 64, 256), 256, 0, stream>>>(ei, alpha, denom, feats, agg);
        k_hnew<<<cdiv(NN * HH, 256), 256, 0, stream>>>(agg, h_new, hnew_bf);

        // evolution gate (WMMA GEMMs)
        gemm(hnew_bf, evoW1t, t_gemm, HH, 128);
        k_evo1post<<<cdiv(NN * 128, 256), 256, 0, stream>>>(t_gemm, evob1, z1_bf);
        gemm(z1_bf, evoW2t, t_gemm, 128, HH);
        k_gate<<<cdiv(NN * HH, 256), 256, 0, stream>>>(t_gemm, evob2, h_new, h_f32, hr, res);
        k_ln_plain<<<NN, 64, 0, stream>>>(hr, h_f32, h_bf);
    }

    // ---- decoders ----
    k_final<<<NN, 64, 0, stream>>>(res, dsW, dsb, diW, dib, dcW, dcb, deW, deb, out);
}
